// VectorQuantizer_67413806678466
// MI455X (gfx1250) — matrix operand order repaired and checked by reading, not compiled, not run on hardware
//
#include <hip/hip_runtime.h>

typedef __attribute__((ext_vector_type(8)))  float  v8f;
typedef __attribute__((ext_vector_type(16))) __bf16 v16bf;
typedef __attribute__((ext_vector_type(8)))  __bf16 v8bf;

#define N_TOK   32768      // 32*32*32 tokens
#define CDIM    256
#define KCB     1024
#define NUMEL   8388608    // 32*256*32*32
// d_out layout: [loss][q_out NUMEL][perplexity][idx 32768]
#define OUT_Q    1
#define OUT_PERP (1 + NUMEL)
#define OUT_IDX  (2 + NUMEL)
// workspace layout (byte offsets)
#define WS_LOSS   0
#define WS_COUNTS 256
#define WS_CBNORM 8192
#define WS_CBHI   16384
#define WS_CBLO   (WS_CBHI + KCB*CDIM*2)
#define WS_XHI    (WS_CBLO + KCB*CDIM*2)
#define WS_XLO    (WS_XHI + N_TOK*CDIM*2)
#define WS_IDX    (WS_XLO + N_TOK*CDIM*2)

__device__ __forceinline__ unsigned short bf16_rne(float f) {
    unsigned u = __float_as_uint(f);
    unsigned r = u + 0x7FFFu + ((u >> 16) & 1u);
    return (unsigned short)(r >> 16);
}
__device__ __forceinline__ float bf16_to_f32(unsigned short h) {
    return __uint_as_float(((unsigned)h) << 16);
}

// CDNA5 async DMA: global -> LDS, 16B per lane, tracked by ASYNCcnt
__device__ __forceinline__ void async_b128(void* lds, const void* g) {
    unsigned l = (unsigned)(unsigned long long)lds;   // LDS offset = low 32 bits
    asm volatile("global_load_async_to_lds_b128 %0, %1, off"
                 :: "v"(l), "v"(g) : "memory");
}
__device__ __forceinline__ void wait_async0() {
    asm volatile("s_wait_asynccnt 0" ::: "memory");
}

// ---------------------------------------------------------------- init
__global__ void vq_init(unsigned* __restrict__ counts, float* __restrict__ loss) {
    int g = blockIdx.x * 256 + threadIdx.x;
    if (g < KCB) counts[g] = 0u;
    if (g == 0) *loss = 0.f;
}

// ------------------------------------------- codebook: norms + bf16 hi/lo
__global__ __launch_bounds__(256)
void vq_cb_prep(const float* __restrict__ cb, unsigned short* __restrict__ cbhi,
                unsigned short* __restrict__ cblo, float* __restrict__ cbnorm) {
    int tid = threadIdx.x, lane = tid & 31, wid = tid >> 5;
    int row = blockIdx.x * 8 + wid;
    const float* r = cb + (size_t)row * CDIM;
    float s = 0.f;
#pragma unroll
    for (int i = 0; i < 8; i++) {
        int c = lane + i * 32;
        float v = r[c];
        s += v * v;
        unsigned short hi = bf16_rne(v);
        unsigned short lo = bf16_rne(v - bf16_to_f32(hi));
        cbhi[(size_t)row * CDIM + c] = hi;
        cblo[(size_t)row * CDIM + c] = lo;
    }
    for (int off = 16; off; off >>= 1) s += __shfl_xor(s, off, 32);
    if (lane == 0) cbnorm[row] = s;
}

// ------------------------- x: NCHW -> [n][c] transpose + bf16 hi/lo split
__global__ __launch_bounds__(256)
void vq_x_to_bf16(const float* __restrict__ x, unsigned short* __restrict__ xhi,
                  unsigned short* __restrict__ xlo) {
    __shared__ float tile[32][33];
    int bid = blockIdx.x;
    int b = bid >> 8;
    int cblk = (bid >> 5) & 7;
    int hwblk = bid & 31;
    int tid = threadIdx.x;
#pragma unroll
    for (int i = 0; i < 4; i++) {
        int e = tid + i * 256;
        int cl = e >> 5, nl = e & 31;
        tile[cl][nl] = x[(size_t)b * 262144 + (size_t)(cblk * 32 + cl) * 1024 + hwblk * 32 + nl];
    }
    __syncthreads();
#pragma unroll
    for (int i = 0; i < 4; i++) {
        int e = tid + i * 256;
        int nl = e >> 5, cl = e & 31;
        float f = tile[cl][nl];
        unsigned short hi = bf16_rne(f);
        unsigned short lo = bf16_rne(f - bf16_to_f32(hi));
        size_t n = (size_t)b * 1024 + hwblk * 32 + nl;
        xhi[n * CDIM + cblk * 32 + cl] = hi;
        xlo[n * CDIM + cblk * 32 + cl] = lo;
    }
}

// --------------------------- main: bf16x3 WMMA GEMM + fused row-argmin
// Grid: 256 blocks (128 rows each), 256 threads = 8 waves (4 along M x 2 along N).
// Double-buffered LDS staging via async global->LDS DMA (ASYNCcnt), overlapped
// with the 24 WMMAs of the previous k-step. One barrier per k-step.
__global__ __launch_bounds__(256)
void vq_argmin_gemm(const unsigned short* __restrict__ xhi, const unsigned short* __restrict__ xlo,
                    const unsigned short* __restrict__ cbhi, const unsigned short* __restrict__ cblo,
                    const float* __restrict__ cbnorm, int* __restrict__ ws_idx,
                    unsigned* __restrict__ counts, float* __restrict__ out) {
    __shared__ unsigned short As_h[2][128][40];   // stride 80B -> bank-spread b128 reads
    __shared__ unsigned short As_l[2][128][40];
    __shared__ unsigned short Bs_h[2][128][40];
    __shared__ unsigned short Bs_l[2][128][40];
    __shared__ float red_v[128][2];
    __shared__ int   red_i[128][2];

    const int tid = threadIdx.x;
    const int lane = tid & 31;
    const int wid = tid >> 5;
    const int waveM = wid & 3;    // 4 waves along M (32 rows each)
    const int waveN = wid >> 2;   // 2 waves along N (64 cols each)
    const int half = lane >> 4;
    const int l15 = lane & 15;
    const int m0 = blockIdx.x * 128;

    // per-thread staging coordinates (2 x b128 per array per step)
    const int r0 = tid >> 2;                 // 0..63
    const int r1 = (tid + 256) >> 2;         // 64..127
    const int kb0 = (tid & 3) << 3;          // 0,8,16,24

    float bestv[2][8];
    int   besti[2][8];
#pragma unroll
    for (int s = 0; s < 2; s++)
#pragma unroll
        for (int r = 0; r < 8; r++) { bestv[s][r] = 3.4e38f; besti[s][r] = 0; }

    union F16 { v16bf v; v8bf h[2]; };
    v8f acc[8];

    // stage(step): async DMA one 128x32 hi/lo tile of A and B into buffer step&1
    auto stage = [&](int step) {
        int nc = step >> 3, ks = step & 7, bf = step & 1;
        int k0 = ks * 32, ncb0 = nc * 128;
        async_b128(&As_h[bf][r0][kb0], xhi + (size_t)(m0 + r0) * CDIM + k0 + kb0);
        async_b128(&As_l[bf][r0][kb0], xlo + (size_t)(m0 + r0) * CDIM + k0 + kb0);
        async_b128(&Bs_h[bf][r0][kb0], cbhi + (size_t)(ncb0 + r0) * CDIM + k0 + kb0);
        async_b128(&Bs_l[bf][r0][kb0], cblo + (size_t)(ncb0 + r0) * CDIM + k0 + kb0);
        async_b128(&As_h[bf][r1][kb0], xhi + (size_t)(m0 + r1) * CDIM + k0 + kb0);
        async_b128(&As_l[bf][r1][kb0], xlo + (size_t)(m0 + r1) * CDIM + k0 + kb0);
        async_b128(&Bs_h[bf][r1][kb0], cbhi + (size_t)(ncb0 + r1) * CDIM + k0 + kb0);
        async_b128(&Bs_l[bf][r1][kb0], cblo + (size_t)(ncb0 + r1) * CDIM + k0 + kb0);
    };

    stage(0);
    for (int step = 0; step < 64; ++step) {       // (nc 0..7) x (ks 0..7)
        const int ks = step & 7;
        const int bf = step & 1;

        wait_async0();                            // my DMA for this step landed
        __syncthreads();                          // everyone's landed; prev reads done
        if (step + 1 < 64) stage(step + 1);       // prefetch next into other buffer

        if (ks == 0) {
#pragma unroll
            for (int i = 0; i < 8; i++) acc[i] = (v8f)(0.0f);
        }

        F16 ah[2], al[2], bh[4], bl[4];
#pragma unroll
        for (int s = 0; s < 2; s++) {             // A frags: 16x32 bf16 layout
            int row = waveM * 32 + s * 16 + l15;
            int kb = half * 8;                    // lanes 16-31 hold K 8..15 / 24..31
            ah[s].h[0] = *(const v8bf*)&As_h[bf][row][kb];
            ah[s].h[1] = *(const v8bf*)&As_h[bf][row][kb + 16];
            al[s].h[0] = *(const v8bf*)&As_l[bf][row][kb];
            al[s].h[1] = *(const v8bf*)&As_l[bf][row][kb + 16];
        }
#pragma unroll
        for (int t = 0; t < 4; t++) {             // B frags: 32x16 bf16 layout
            int col = waveN * 64 + t * 16 + l15;
            int off = half * 8;                   // lanes 16-31 hold K 8..15 / 24..31
            bh[t].h[0] = *(const v8bf*)&Bs_h[bf][col][off];
            bh[t].h[1] = *(const v8bf*)&Bs_h[bf][col][off + 16];
            bl[t].h[0] = *(const v8bf*)&Bs_l[bf][col][off];
            bl[t].h[1] = *(const v8bf*)&Bs_l[bf][col][off + 16];
        }
#pragma unroll
        for (int s = 0; s < 2; s++)
#pragma unroll
            for (int t = 0; t < 4; t++) {         // bf16x3: hh + lh + hl
                int a = s * 4 + t;
                acc[a] = __builtin_amdgcn_wmma_f32_16x16x32_bf16(false, ah[s].v, false, bh[t].v,
                                                                 (short)0, acc[a], false, false);
                acc[a] = __builtin_amdgcn_wmma_f32_16x16x32_bf16(false, al[s].v, false, bh[t].v,
                                                                 (short)0, acc[a], false, false);
                acc[a] = __builtin_amdgcn_wmma_f32_16x16x32_bf16(false, ah[s].v, false, bl[t].v,
                                                                 (short)0, acc[a], false, false);
            }

        if (ks == 7) {                            // chunk epilogue: running argmin
            const int ncb0 = (step >> 3) * 128;
#pragma unroll
            for (int t = 0; t < 4; t++) {
                int col = ncb0 + waveN * 64 + t * 16 + l15;
                float cn = cbnorm[col];           // dist = ||e||^2 - 2*dot
#pragma unroll
                for (int s = 0; s < 2; s++)
#pragma unroll
                    for (int r = 0; r < 8; r++) {
                        float d = cn - 2.0f * acc[s * 4 + t][r];
                        if (d < bestv[s][r]) { bestv[s][r] = d; besti[s][r] = col; }
                    }
            }
        }
    }

    // cross-lane (16-lane halves) lexicographic min, then cross-wave via LDS
#pragma unroll
    for (int s = 0; s < 2; s++)
#pragma unroll
        for (int r = 0; r < 8; r++) {
            float v = bestv[s][r];
            int bi = besti[s][r];
            for (int off = 1; off < 16; off <<= 1) {
                float ov = __shfl_xor(v, off, 32);
                int oi = __shfl_xor(bi, off, 32);
                if (ov < v || (ov == v && oi < bi)) { v = ov; bi = oi; }
            }
            if (l15 == 0) {                       // lanes 0 and 16: two row halves
                int rowl = waveM * 32 + s * 16 + half * 8 + r;
                red_v[rowl][waveN] = v;
                red_i[rowl][waveN] = bi;
            }
        }
    __syncthreads();
    if (tid < 128) {
        float v0 = red_v[tid][0]; int i0 = red_i[tid][0];
        float v1 = red_v[tid][1]; int i1 = red_i[tid][1];
        int bi = (v1 < v0 || (v1 == v0 && i1 < i0)) ? i1 : i0;
        int m = m0 + tid;
        ws_idx[m] = bi;
        atomicAdd(&counts[bi], 1u);
        out[OUT_IDX + m] = (float)bi;
    }
}

// ------------------- gather in f32, write q_out (NCHW), accumulate loss
__global__ __launch_bounds__(256)
void vq_output(const float* __restrict__ x, const float* __restrict__ cb,
               const int* __restrict__ ws_idx, float* __restrict__ out,
               float* __restrict__ loss) {
    __shared__ int idxs[32];
    __shared__ float part[8];
    int tid = threadIdx.x;
    int b = blockIdx.x >> 5, h = blockIdx.x & 31;
    if (tid < 32) idxs[tid] = ws_idx[b * 1024 + h * 32 + tid];
    __syncthreads();
    int w = tid & 31, c0 = tid >> 5;
    int k = idxs[w];
    float local = 0.f;
#pragma unroll
    for (int i = 0; i < 32; ++i) {
        int c = c0 + i * 8;
        float q = cb[(size_t)k * CDIM + c];
        size_t gi = (size_t)b * 262144 + (size_t)c * 1024 + h * 32 + w;
        float d = q - x[gi];
        local += d * d;
        out[OUT_Q + gi] = q;                      // quantized_st == quantized
    }
    for (int off = 16; off; off >>= 1) local += __shfl_xor(local, off, 32);
    if ((tid & 31) == 0) part[tid >> 5] = local;
    __syncthreads();
    if (tid == 0) {
        float s = 0.f;
        for (int i = 0; i < 8; i++) s += part[i];
        atomicAdd(loss, s);
    }
}

// ------------------------------------------------ perplexity + loss scalar
__global__ __launch_bounds__(1024)
void vq_finalize(const unsigned* __restrict__ counts, const float* __restrict__ loss,
                 float* __restrict__ out) {
    __shared__ float part[32];
    int tid = threadIdx.x;
    float p = (float)counts[tid] * (1.0f / 32768.0f);
    float t = -p * logf(p + 1e-10f);
    for (int off = 16; off; off >>= 1) t += __shfl_xor(t, off, 32);
    if ((tid & 31) == 0) part[tid >> 5] = t;
    __syncthreads();
    if (tid == 0) {
        float H = 0.f;
        for (int i = 0; i < 32; i++) H += part[i];
        out[OUT_PERP] = expf(H);
        out[0] = (*loss) * 1.25f / 8388608.0f;    // q_latent + 0.25*e_latent
    }
}

extern "C" void kernel_launch(void* const* d_in, const int* in_sizes, int n_in,
                              void* d_out, int out_size, void* d_ws, size_t ws_size,
                              hipStream_t stream) {
    const float* x  = (const float*)d_in[0];   // [32,256,32,32]
    const float* cb = (const float*)d_in[1];   // [1024,256]
    float* out = (float*)d_out;
    char* ws = (char*)d_ws;

    float*          loss   = (float*)(ws + WS_LOSS);
    unsigned*       counts = (unsigned*)(ws + WS_COUNTS);
    float*          cbnorm = (float*)(ws + WS_CBNORM);
    unsigned short* cbhi   = (unsigned short*)(ws + WS_CBHI);
    unsigned short* cblo   = (unsigned short*)(ws + WS_CBLO);
    unsigned short* xhi    = (unsigned short*)(ws + WS_XHI);
    unsigned short* xlo    = (unsigned short*)(ws + WS_XLO);
    int*            idxbuf = (int*)(ws + WS_IDX);

    vq_init<<<4, 256, 0, stream>>>(counts, loss);
    vq_cb_prep<<<128, 256, 0, stream>>>(cb, cbhi, cblo, cbnorm);
    vq_x_to_bf16<<<8192, 256, 0, stream>>>(x, xhi, xlo);
    vq_argmin_gemm<<<256, 256, 0, stream>>>(xhi, xlo, cbhi, cblo, cbnorm, idxbuf, counts, out);
    vq_output<<<1024, 256, 0, stream>>>(x, cb, idxbuf, out, loss);
    vq_finalize<<<1, 1024, 0, stream>>>(counts, loss, out);
}
